// TernaryCNN_33586644255324
// MI455X (gfx1250) — matrix operand order repaired and checked by reading, not compiled, not run on hardware
//
#include <hip/hip_runtime.h>
#include <cstddef>
#include <cstdint>

typedef __attribute__((ext_vector_type(16))) _Float16 v16h;
typedef __attribute__((ext_vector_type(8)))  _Float16 v8h;
typedef __attribute__((ext_vector_type(8)))  float    v8f;

__device__ __forceinline__ v16h cat8(v8h lo, v8h hi) {
    return __builtin_shufflevector(lo, hi, 0,1,2,3,4,5,6,7,8,9,10,11,12,13,14,15);
}
__device__ __forceinline__ v8f wmma_f16(v16h a, v16h b, v8f c) {
    return __builtin_amdgcn_wmma_f32_16x16x32_f16(false, a, false, b, (short)0, c, false, false);
}

// ---------------------------------------------------------------------------
// Ternarization: 3-pass TWN (delta = 0.7*mean|w|, alpha = mean |w| over nz)
// ---------------------------------------------------------------------------
__global__ __launch_bounds__(256) void absum_kernel(const float* __restrict__ w, long n,
                                                    double* __restrict__ out) {
    __shared__ double sm[256];
    double s = 0.0;
    for (long i = (long)blockIdx.x * 256 + threadIdx.x; i < n; i += (long)gridDim.x * 256)
        s += (double)fabsf(w[i]);
    sm[threadIdx.x] = s; __syncthreads();
    for (int o = 128; o > 0; o >>= 1) {
        if ((int)threadIdx.x < o) sm[threadIdx.x] += sm[threadIdx.x + o];
        __syncthreads();
    }
    if (threadIdx.x == 0) atomicAdd(out, sm[0]);
}

__global__ __launch_bounds__(256) void nzstat_kernel(const float* __restrict__ w, long n,
                                                     const double* __restrict__ sumabs,
                                                     double* __restrict__ nzsum,
                                                     unsigned long long* __restrict__ cnt) {
    __shared__ double sm[256];
    __shared__ unsigned long long cm[256];
    const double delta = 0.7 * (*sumabs) / (double)n;
    double s = 0.0; unsigned long long c = 0;
    for (long i = (long)blockIdx.x * 256 + threadIdx.x; i < n; i += (long)gridDim.x * 256) {
        float a = fabsf(w[i]);
        if ((double)a > delta) { s += (double)a; ++c; }
    }
    sm[threadIdx.x] = s; cm[threadIdx.x] = c; __syncthreads();
    for (int o = 128; o > 0; o >>= 1) {
        if ((int)threadIdx.x < o) { sm[threadIdx.x] += sm[threadIdx.x + o]; cm[threadIdx.x] += cm[threadIdx.x + o]; }
        __syncthreads();
    }
    if (threadIdx.x == 0) { atomicAdd(nzsum, sm[0]); atomicAdd(cnt, cm[0]); }
}

// Write f16 ternary codes {-1,0,+1} into K-padded [rows][Kpad] layout.
// grid.y = row, grid.x covers Kpad -> no runtime division.
__global__ __launch_bounds__(256) void tern_write_kernel(const float* __restrict__ w,
                                                         int Kt, int Kpad, long n_elems,
                                                         const double* __restrict__ sumabs,
                                                         const double* __restrict__ nzsum,
                                                         const unsigned long long* __restrict__ cnt,
                                                         double* __restrict__ alpha_out,
                                                         _Float16* __restrict__ out) {
    const double sa = *sumabs;
    const double delta = 0.7 * sa / (double)n_elems;
    const int row = blockIdx.y;
    const int k = blockIdx.x * 256 + threadIdx.x;
    if (k < Kpad) {
        _Float16 v = (_Float16)0;
        if (k < Kt) {
            float xw = w[(long)row * Kt + k];
            if ((double)xw >  delta) v = (_Float16)1.0f;
            else if ((double)xw < -delta) v = (_Float16)(-1.0f);
        }
        out[(long)row * Kpad + k] = v;
    }
    if (blockIdx.x == 0 && blockIdx.y == 0 && threadIdx.x == 0) {
        const unsigned long long c = *cnt;
        *alpha_out = c ? (*nzsum) / (double)c : sa / (double)n_elems;
    }
}

__global__ __launch_bounds__(256) void f32_to_f16_kernel(const float* __restrict__ in,
                                                         _Float16* __restrict__ out, long n) {
    for (long i = (long)blockIdx.x * 256 + threadIdx.x; i < n; i += (long)gridDim.x * 256)
        out[i] = (_Float16)in[i];
}

// ---------------------------------------------------------------------------
// Implicit-GEMM conv via WMMA. Wave tile: 16 (cout) x 64 (pixels).
// grid.x = P/512 (8 pixel tiles per block, one per wave), grid.y = Cout/16.
// W,H are powers of two -> pixel decode is shift/mask only.
// B (im2col) staged transposed in LDS [pixel][k]; A is unguarded b128 loads
// from the K-padded ternary weight layout. Epilogue: alpha*acc + bias -> f32.
// ---------------------------------------------------------------------------
__global__ __launch_bounds__(256) void conv_wmma_kernel(const _Float16* __restrict__ xin,
                                                        const _Float16* __restrict__ wt,
                                                        const float* __restrict__ bias,
                                                        const double* __restrict__ alpha_p,
                                                        float* __restrict__ ypre,
                                                        int Cin, int Cout,
                                                        int lW, int lH) {
    __shared__ __align__(32) _Float16 lds[8 * 64 * 32];   // 32 KB: per-wave 64px x 32k
    const int W = 1 << lW, H = 1 << lH;
    const int KT    = Cin * 9;
    const int Kpad  = (KT + 31) & ~31;
    const int lane  = threadIdx.x & 31;
    const int wave  = threadIdx.x >> 5;
    const int pcol  = lane & 15;
    const int khalf = lane >> 4;
    const int mt    = blockIdx.y;
    const long p0   = ((long)blockIdx.x * 8 + wave) * 64;

    _Float16* myb = lds + wave * (64 * 32);

    int gn[4], gh[4], gw0[4];
#pragma unroll
    for (int g = 0; g < 4; ++g) {
        long pix = p0 + (long)g * 16;                      // 16 pixels, same row (W%16==0)
        gw0[g] = (int)(pix & (W - 1));
        gh[g]  = (int)((pix >> lW) & (H - 1));
        gn[g]  = (int)(pix >> (lW + lH));
    }

    v8f acc[4] = {v8f{}, v8f{}, v8f{}, v8f{}};

    for (int kk = 0; kk < KT; kk += 32) {
        __syncthreads();
        {   // stage: lane owns k = kk+lane for all 64 pixels; transposed store
            int kidx = kk + lane;
            bool kok = kidx < KT;
            int ci = kidx / 9, rr = kidx % 9;              // constant divisor: mul-shift
            int kh = rr / 3, kw = rr % 3;
#pragma unroll
            for (int g = 0; g < 4; ++g) {
                int hin = gh[g] + kh - 1;
                bool hok = kok && (hin >= 0) && (hin < H);
                int wbase = gw0[g] + kw - 1;
                const _Float16* src = xin +
                    ((((long)gn[g] * Cin + (hok ? ci : 0)) << (lW + lH)) + ((long)(hok ? hin : 0) << lW)) + wbase;
                if (hok && wbase >= 0 && (wbase + 16) <= W) {
                    // fast interior path: unguarded contiguous loads
#pragma unroll
                    for (int j = 0; j < 16; ++j)
                        myb[(g * 16 + j) * 32 + lane] = src[j];
                } else {
#pragma unroll 1
                    for (int j = 0; j < 16; ++j) {
                        int wj = wbase + j;
                        _Float16 v = (_Float16)0;
                        if (hok && (unsigned)wj < (unsigned)W) v = src[j];
                        myb[(g * 16 + j) * 32 + lane] = v;
                    }
                }
            }
        }
        __syncthreads();
        // A fragment: lanes 0-15 hold K 0-15, lanes 16-31 hold K 16-31
        const _Float16* wrow = wt + (long)(mt * 16 + pcol) * Kpad + kk + khalf * 16;
        v16h a = cat8(*(const v8h*)wrow, *(const v8h*)(wrow + 8));
#pragma unroll
        for (int g = 0; g < 4; ++g) {
            v16h b = *(const v16h*)&myb[(g * 16 + pcol) * 32 + khalf * 16];
            acc[g] = wmma_f16(a, b, acc[g]);
        }
    }

    const float al = (float)(*alpha_p);
#pragma unroll
    for (int g = 0; g < 4; ++g) {
        int n = gn[g], h = gh[g], w = gw0[g] + pcol;
#pragma unroll
        for (int r = 0; r < 8; ++r) {
            int co = mt * 16 + r + 8 * khalf;
            ypre[((((long)n * Cout + co) << lH) + h << lW) + w] = al * acc[g][r] + bias[co];
        }
    }
}

// ---------------------------------------------------------------------------
// BatchNorm training-mode stats: one block per channel -> scale/shift.
// HW is a power of two -> shift/mask decode.
// ---------------------------------------------------------------------------
__global__ __launch_bounds__(256) void bn_stats_kernel(const float* __restrict__ y,
                                                       const float* __restrict__ gamma,
                                                       const float* __restrict__ beta,
                                                       float* __restrict__ scale,
                                                       float* __restrict__ shift,
                                                       int N, int C, int lHW) {
    __shared__ double s1[256], s2[256];
    const int c = blockIdx.x;
    const long HW = 1L << lHW;
    const long cnt = (long)N << lHW;
    double a = 0.0, b = 0.0;
    for (long i = threadIdx.x; i < cnt; i += 256) {
        long nn = i >> lHW, r = i & (HW - 1);
        float v = y[(((long)nn * C + c) << lHW) + r];
        a += (double)v; b += (double)v * (double)v;
    }
    s1[threadIdx.x] = a; s2[threadIdx.x] = b; __syncthreads();
    for (int o = 128; o > 0; o >>= 1) {
        if ((int)threadIdx.x < o) { s1[threadIdx.x] += s1[threadIdx.x + o]; s2[threadIdx.x] += s2[threadIdx.x + o]; }
        __syncthreads();
    }
    if (threadIdx.x == 0) {
        double mean = s1[0] / (double)cnt;
        double var  = s2[0] / (double)cnt - mean * mean;
        float sc = gamma[c] * rsqrtf((float)var + 1e-5f);
        scale[c] = sc;
        shift[c] = beta[c] - (float)mean * sc;
    }
}

// Fused BN-affine + ReLU + 2x2 maxpool -> f16 activations for next layer.
// All of Wo, Ho, C are powers of two -> shift/mask decode.
__global__ __launch_bounds__(256) void bn_pool_kernel(const float* __restrict__ y,
                                                      const float* __restrict__ scale,
                                                      const float* __restrict__ shift,
                                                      _Float16* __restrict__ out,
                                                      int N, int lC, int lH, int lW) {
    const int lHo = lH - 1, lWo = lW - 1;
    const int W = 1 << lW;
    const long total = ((long)N << lC) << (lHo + lWo);
    for (long i = (long)blockIdx.x * 256 + threadIdx.x; i < total; i += (long)gridDim.x * 256) {
        int ow = (int)(i & ((1 << lWo) - 1));
        int oh = (int)((i >> lWo) & ((1 << lHo) - 1));
        int c  = (int)((i >> (lWo + lHo)) & ((1 << lC) - 1));
        int n  = (int)(i >> (lWo + lHo + lC));
        const float* p = y + ((((long)n << lC) + c) << (lH + lW)) + ((long)(2 * oh) << lW) + 2 * ow;
        float sc = scale[c], sh = shift[c];
        float a0 = p[0] * sc + sh, a1 = p[1] * sc + sh;
        float a2 = p[W] * sc + sh, a3 = p[W + 1] * sc + sh;
        float m = fmaxf(fmaxf(a0, a1), fmaxf(a2, a3));
        out[i] = (_Float16)fmaxf(m, 0.0f);
    }
}

// ---------------------------------------------------------------------------
// FC via WMMA: wave tile 32(batch) x 16(out-features); two accumulators share
// one B fragment. K multiple of 32 -> unguarded contiguous fragment loads.
// ---------------------------------------------------------------------------
__global__ __launch_bounds__(256) void fc_wmma_kernel(const _Float16* __restrict__ act,  // [32][K]
                                                      const _Float16* __restrict__ wt,   // [Nfull][K]
                                                      const float* __restrict__ bias,
                                                      const double* __restrict__ alpha_p,
                                                      _Float16* __restrict__ out_h,
                                                      float* __restrict__ out_f,
                                                      int K, int Nfull, int Nstride, int do_relu) {
    const int lane  = threadIdx.x & 31;
    const int wave  = threadIdx.x >> 5;
    const int wtile = blockIdx.x * 8 + wave;
    const int ntiles = (Nfull + 15) >> 4;
    if (wtile >= ntiles) return;                  // no barriers below: safe
    const int pcol = lane & 15, khalf = lane >> 4;
    const int n = wtile * 16 + pcol;
    const bool nok = n < Nfull;

    const _Float16* a0row = act + (long)pcol * K;
    const _Float16* a1row = act + (long)(pcol + 16) * K;
    const _Float16* brow  = wt + (long)(nok ? n : 0) * K;

    v8f acc0 = {}, acc1 = {};
    for (int kk = 0; kk < K; kk += 32) {
        v16h a0 = cat8(*(const v8h*)(a0row + kk + khalf * 16), *(const v8h*)(a0row + kk + khalf * 16 + 8));
        v16h a1 = cat8(*(const v8h*)(a1row + kk + khalf * 16), *(const v8h*)(a1row + kk + khalf * 16 + 8));
        v16h b = {};
        if (nok) b = *(const v16h*)(brow + kk + khalf * 16);
        acc0 = wmma_f16(a0, b, acc0);
        acc1 = wmma_f16(a1, b, acc1);
    }
    if (!nok) return;
    const float al = (float)(*alpha_p);
    const float bn = bias[n];
#pragma unroll
    for (int r = 0; r < 8; ++r) {
        int m0 = r + 8 * khalf;
        float v0 = al * acc0[r] + bn;
        float v1 = al * acc1[r] + bn;
        if (do_relu) {
            out_h[(long)m0 * Nstride + n]        = (_Float16)fmaxf(v0, 0.0f);
            out_h[(long)(m0 + 16) * Nstride + n] = (_Float16)fmaxf(v1, 0.0f);
        } else {
            out_f[(long)m0 * Nstride + n]        = v0;
            out_f[(long)(m0 + 16) * Nstride + n] = v1;
        }
    }
}

// ---------------------------------------------------------------------------
static inline int grid1(long n) { long b = (n + 255) / 256; return (int)(b > 4096 ? 4096 : (b < 1 ? 1 : b)); }

extern "C" void kernel_launch(void* const* d_in, const int* in_sizes, int n_in,
                              void* d_out, int out_size, void* d_ws, size_t ws_size,
                              hipStream_t stream) {
    const float* x = (const float*)d_in[0];
    const float* w[4]  = {(const float*)d_in[1], (const float*)d_in[5], (const float*)d_in[9],  (const float*)d_in[13]};
    const float* bi[4] = {(const float*)d_in[2], (const float*)d_in[6], (const float*)d_in[10], (const float*)d_in[14]};
    const float* ga[4] = {(const float*)d_in[3], (const float*)d_in[7], (const float*)d_in[11], (const float*)d_in[15]};
    const float* be[4] = {(const float*)d_in[4], (const float*)d_in[8], (const float*)d_in[12], (const float*)d_in[16]};
    const float* fw1 = (const float*)d_in[17];
    const float* fb1 = (const float*)d_in[18];
    const float* fw2 = (const float*)d_in[19];
    const float* fb2 = (const float*)d_in[20];
    float* out = (float*)d_out;
    char* ws = (char*)d_ws;

    size_t off = 0;
    auto alloc = [&](size_t b) { size_t o = off; off = (off + b + 255) & ~(size_t)255; return o; };

    struct TInfo { const float* w; long rows, Kt, Kpad; };
    TInfo tern[6] = {
        {w[0],   64,    27,    32},
        {w[1],  128,   576,   576},
        {w[2],  256,  1152,  1152},
        {w[3],  512,  2304,  2304},
        {fw1,  1024, 32768, 32768},
        {fw2,  1000,  1024,  1024},
    };

    const size_t o_stats = alloc(1024);
    const size_t o_bn    = alloc(1024 * sizeof(float));      // scale[512] + shift[512]
    size_t o_wt[6];
    for (int t = 0; t < 6; ++t) o_wt[t] = alloc((size_t)tern[t].rows * tern[t].Kpad * 2);
    const size_t o_xh   = alloc((size_t)32 * 3 * 128 * 128 * 2);
    const size_t o_actA = alloc((size_t)32 * 64 * 64 * 64 * 2);
    const size_t o_actB = alloc((size_t)32 * 64 * 64 * 64 * 2);
    const size_t o_ypre = alloc((size_t)32 * 64 * 128 * 128 * 4);
    const size_t o_fc1  = alloc((size_t)32 * 1024 * 2);
    (void)ws_size; (void)n_in; (void)in_sizes; (void)out_size;

    double* sums  = (double*)(ws + o_stats);                          // [2*t], [2*t+1]
    unsigned long long* cnts = (unsigned long long*)(ws + o_stats + 96);
    double* alphas = (double*)(ws + o_stats + 160);
    float* bnscale = (float*)(ws + o_bn);
    float* bnshift = bnscale + 512;

    hipMemsetAsync(ws + o_stats, 0, 1024, stream);

    // --- ternarize all 6 weight tensors ---
    for (int t = 0; t < 6; ++t) {
        long n = tern[t].rows * tern[t].Kt;
        absum_kernel<<<grid1(n), 256, 0, stream>>>(tern[t].w, n, &sums[2 * t]);
        nzstat_kernel<<<grid1(n), 256, 0, stream>>>(tern[t].w, n, &sums[2 * t], &sums[2 * t + 1], &cnts[t]);
        dim3 g2((unsigned)((tern[t].Kpad + 255) / 256), (unsigned)tern[t].rows);
        tern_write_kernel<<<g2, 256, 0, stream>>>(tern[t].w, (int)tern[t].Kt, (int)tern[t].Kpad, n,
            &sums[2 * t], &sums[2 * t + 1], &cnts[t], &alphas[t], (_Float16*)(ws + o_wt[t]));
    }

    // --- input to f16 ---
    {
        long n = (long)32 * 3 * 128 * 128;
        f32_to_f16_kernel<<<grid1(n), 256, 0, stream>>>(x, (_Float16*)(ws + o_xh), n);
    }

    // --- conv blocks (ping-pong activations) ---
    struct LInfo { int Cin, Cout, lH, lW; size_t in_off, out_off; };
    LInfo L[4] = {
        {  3,  64, 7, 7, o_xh,   o_actA},
        { 64, 128, 6, 6, o_actA, o_actB},
        {128, 256, 5, 5, o_actB, o_actA},
        {256, 512, 4, 4, o_actA, o_actB},
    };
    for (int l = 0; l < 4; ++l) {
        const int N = 32, Cin = L[l].Cin, Cout = L[l].Cout, lH = L[l].lH, lW = L[l].lW;
        const long P = (long)N << (lH + lW);
        dim3 cg((unsigned)(P / 512), (unsigned)(Cout / 16));
        conv_wmma_kernel<<<cg, 256, 0, stream>>>(
            (const _Float16*)(ws + L[l].in_off), (const _Float16*)(ws + o_wt[l]),
            bi[l], &alphas[l], (float*)(ws + o_ypre), Cin, Cout, lW, lH);
        bn_stats_kernel<<<Cout, 256, 0, stream>>>((const float*)(ws + o_ypre), ga[l], be[l],
                                                  bnscale, bnshift, N, Cout, lH + lW);
        long tot = ((long)N * Cout) << (lH + lW - 2);
        int lC = (Cout == 64) ? 6 : (Cout == 128) ? 7 : (Cout == 256) ? 8 : 9;
        bn_pool_kernel<<<grid1(tot), 256, 0, stream>>>((const float*)(ws + o_ypre), bnscale, bnshift,
                                                       (_Float16*)(ws + L[l].out_off), N, lC, lH, lW);
    }

    // --- FC1: [32,32768] x [1024,32768]^T, relu, f16 out; ntiles=64 -> 8 blocks ---
    fc_wmma_kernel<<<8, 256, 0, stream>>>((const _Float16*)(ws + o_actB), (const _Float16*)(ws + o_wt[4]),
                                          fb1, &alphas[4], (_Float16*)(ws + o_fc1), nullptr,
                                          32768, 1024, 1024, 1);
    // --- FC2: [32,1024] x [1000,1024]^T, f32 out to d_out; ntiles=63 -> 8 blocks ---
    fc_wmma_kernel<<<8, 256, 0, stream>>>((const _Float16*)(ws + o_fc1), (const _Float16*)(ws + o_wt[5]),
                                          fb2, &alphas[5], nullptr, out,
                                          1024, 1000, 1000, 0);
}
